// TransformerEncoderLayer_77343771066479
// MI455X (gfx1250) — compile-verified, matrix operand order checked by reading, not run on hardware
//
#include <hip/hip_runtime.h>

// ---------------- types ----------------
typedef __bf16 bf16;
typedef __bf16 v16bf __attribute__((ext_vector_type(16)));
typedef __bf16 bf16x8 __attribute__((ext_vector_type(8)));
typedef float  v8f   __attribute__((ext_vector_type(8)));

#define D_MODEL 1024
#define SEQ     2048
#define BATCH   2
#define M_TOK   (BATCH * SEQ)   // 4096
#define N_HEADS 16
#define D_HEAD  64
#define FFN     4096
#define QKV_N   (3 * D_MODEL)   // 3072
#define NEGV    (-1000000000.0f)

// ---------------- fragment loader ----------------
// Loads a 16x32 (bf16) WMMA operand fragment from a row-major tile.
// base points at (row_base, k_base). Per-lane: row = lane&15,
// chunks at k = (lane>>4)*8 and (lane>>4)*8 + 16  (two 16-byte loads).
__device__ __forceinline__ v16bf ld_frag(const bf16* base, int ld, int lane) {
  const bf16* p = base + (size_t)(lane & 15) * ld + ((lane >> 4) << 3);
  bf16x8 lo = *(const bf16x8*)(p);
  bf16x8 hi = *(const bf16x8*)(p + 16);
  v16bf f;
#pragma unroll
  for (int i = 0; i < 8; ++i) { f[i] = lo[i]; f[i + 8] = hi[i]; }
  return f;
}

__device__ __forceinline__ v8f wmma_bf16(v16bf a, v16bf b, v8f c) {
  return __builtin_amdgcn_wmma_f32_16x16x32_bf16(false, a, false, b,
                                                 (short)0, c, false, false);
}

// ---------------- weight transpose + bf16 convert ----------------
// W: fp32 [K,N] row-major  ->  Wt: bf16 [N,K] row-major
__global__ __launch_bounds__(256) void transw_kernel(
    const float* __restrict__ W, bf16* __restrict__ Wt, int K, int N) {
  __shared__ float tile[32][33];
  int kb = blockIdx.y * 32, nb = blockIdx.x * 32;
  int tx = threadIdx.x & 31, ty = threadIdx.x >> 5;  // 32 x 8
#pragma unroll
  for (int i = 0; i < 32; i += 8)
    tile[ty + i][tx] = W[(size_t)(kb + ty + i) * N + (nb + tx)];
  __syncthreads();
#pragma unroll
  for (int i = 0; i < 32; i += 8)
    Wt[(size_t)(nb + ty + i) * K + (kb + tx)] = (bf16)tile[tx][ty + i];
}

// ---------------- pack 3 bias vectors ----------------
__global__ __launch_bounds__(256) void pack3_kernel(
    const float* __restrict__ a, const float* __restrict__ b,
    const float* __restrict__ c, float* __restrict__ o) {
  int i = blockIdx.x * 256 + threadIdx.x;  // 0..3071
  float v;
  if (i < D_MODEL) v = a[i];
  else if (i < 2 * D_MODEL) v = b[i - D_MODEL];
  else v = c[i - 2 * D_MODEL];
  o[i] = v;
}

// ---------------- block reduction ----------------
__device__ __forceinline__ float block_reduce_sum(float v, float* red) {
  int t = threadIdx.x;
  red[t] = v;
  __syncthreads();
  for (int s = 128; s > 0; s >>= 1) {
    if (t < s) red[t] += red[t + s];
    __syncthreads();
  }
  float r = red[0];
  __syncthreads();
  return r;
}

// ---------------- embed + pos + LN1 ----------------
__global__ __launch_bounds__(256) void embed_ln_kernel(
    const int* __restrict__ tokens, const float* __restrict__ emb,
    const float* __restrict__ pos, const float* __restrict__ g,
    const float* __restrict__ bb, float* __restrict__ xf,
    bf16* __restrict__ xb) {
  __shared__ float red[256];
  int row = blockIdx.x;            // b*SEQ + s
  int s = row & (SEQ - 1);
  int tok = tokens[row];
  float e[4];
  float sum = 0.f;
#pragma unroll
  for (int i = 0; i < 4; ++i) {
    int idx = threadIdx.x + i * 256;
    e[i] = emb[(size_t)tok * D_MODEL + idx] + pos[(size_t)s * D_MODEL + idx];
    sum += e[i];
  }
  sum = block_reduce_sum(sum, red);
  float mean = sum * (1.0f / D_MODEL);
  float vs = 0.f;
#pragma unroll
  for (int i = 0; i < 4; ++i) { float d = e[i] - mean; vs += d * d; }
  vs = block_reduce_sum(vs, red);
  float rstd = rsqrtf(vs * (1.0f / D_MODEL) + 1e-5f);
#pragma unroll
  for (int i = 0; i < 4; ++i) {
    int idx = threadIdx.x + i * 256;
    float y = (e[i] - mean) * rstd * g[idx] + bb[idx];
    xf[(size_t)row * D_MODEL + idx] = y;
    xb[(size_t)row * D_MODEL + idx] = (bf16)y;
  }
}

// ---------------- fused  x2 = x + LN2(attn_out);  hin = bf16(LN3(x2)) ----------------
__global__ __launch_bounds__(256) void ln2ln3_kernel(
    const float* __restrict__ xf, const float* __restrict__ ao,
    const float* __restrict__ g2, const float* __restrict__ b2,
    const float* __restrict__ g3, const float* __restrict__ b3,
    float* __restrict__ x2f, bf16* __restrict__ hin) {
  __shared__ float red[256];
  int row = blockIdx.x;
  float a[4];
  float sum = 0.f;
#pragma unroll
  for (int i = 0; i < 4; ++i) {
    int idx = threadIdx.x + i * 256;
    a[i] = ao[(size_t)row * D_MODEL + idx];
    sum += a[i];
  }
  sum = block_reduce_sum(sum, red);
  float mean = sum * (1.0f / D_MODEL);
  float vs = 0.f;
#pragma unroll
  for (int i = 0; i < 4; ++i) { float d = a[i] - mean; vs += d * d; }
  vs = block_reduce_sum(vs, red);
  float rstd = rsqrtf(vs * (1.0f / D_MODEL) + 1e-5f);
  float x2[4];
  float sum2 = 0.f;
#pragma unroll
  for (int i = 0; i < 4; ++i) {
    int idx = threadIdx.x + i * 256;
    float t = (a[i] - mean) * rstd * g2[idx] + b2[idx];
    x2[i] = xf[(size_t)row * D_MODEL + idx] + t;
    x2f[(size_t)row * D_MODEL + idx] = x2[i];
    sum2 += x2[i];
  }
  sum2 = block_reduce_sum(sum2, red);
  float mean2 = sum2 * (1.0f / D_MODEL);
  float vs2 = 0.f;
#pragma unroll
  for (int i = 0; i < 4; ++i) { float d = x2[i] - mean2; vs2 += d * d; }
  vs2 = block_reduce_sum(vs2, red);
  float rstd2 = rsqrtf(vs2 * (1.0f / D_MODEL) + 1e-5f);
#pragma unroll
  for (int i = 0; i < 4; ++i) {
    int idx = threadIdx.x + i * 256;
    float y = (x2[i] - mean2) * rstd2 * g3[idx] + b3[idx];
    hin[(size_t)row * D_MODEL + idx] = (bf16)y;
  }
}

// ---------------- generic bf16 WMMA GEMM (double-buffered LDS) ----------------
// C[M,N] = A[M,K] (bf16, row-major) * Bt[N,K]^T (bf16, N-major) + bias
// optional exact GELU, optional fp32 residual add; writes fp32 and/or bf16.
__global__ __launch_bounds__(256) void gemm_bf16_kernel(
    const bf16* __restrict__ A, const bf16* __restrict__ Bt,
    const float* __restrict__ bias, const float* __restrict__ resid,
    float* __restrict__ Cf, bf16* __restrict__ Cb,
    int M, int N, int K, int do_gelu) {
  __shared__ bf16 As[2][128 * 40];
  __shared__ bf16 Bs[2][128 * 40];
  const int lane = threadIdx.x & 31;
  const int wave = threadIdx.x >> 5;   // 8 waves
  const int wm = wave & 3;             // 4 strips along M (32 rows)
  const int wn = wave >> 2;            // 2 strips along N (64 cols)
  const int m0 = blockIdx.y * 128;
  const int n0 = blockIdx.x * 128;

  v8f acc[2][4];
#pragma unroll
  for (int a = 0; a < 2; ++a)
#pragma unroll
    for (int b = 0; b < 4; ++b)
#pragma unroll
      for (int r = 0; r < 8; ++r) acc[a][b][r] = 0.f;

  const int srow = threadIdx.x >> 1;        // staging row 0..127
  const int co = (threadIdx.x & 1) * 16;    // staging col offset (16 bf16 = 32B)
  const bf16* ga = A + (size_t)(m0 + srow) * K + co;
  const bf16* gb = Bt + (size_t)(n0 + srow) * K + co;

  // prologue: stage K-tile 0 into buffer 0
  {
    bf16x8 a0 = *(const bf16x8*)(ga);
    bf16x8 a1 = *(const bf16x8*)(ga + 8);
    bf16x8 b0 = *(const bf16x8*)(gb);
    bf16x8 b1 = *(const bf16x8*)(gb + 8);
    *(bf16x8*)&As[0][srow * 40 + co]     = a0;
    *(bf16x8*)&As[0][srow * 40 + co + 8] = a1;
    *(bf16x8*)&Bs[0][srow * 40 + co]     = b0;
    *(bf16x8*)&Bs[0][srow * 40 + co + 8] = b1;
  }

  const int nk = K >> 5;
  for (int i = 0; i < nk; ++i) {
    __syncthreads();  // buffer i&1 visible; all waves done with buffer (i+1)&1

    // prefetch next K-tile from global (overlaps the WMMA chain)
    bf16x8 na0{}, na1{}, nb0{}, nb1{};
    const bool pf = (i + 1) < nk;
    if (pf) {
      const bf16* pga = ga + (size_t)(i + 1) * 32;
      const bf16* pgb = gb + (size_t)(i + 1) * 32;
      na0 = *(const bf16x8*)(pga);
      na1 = *(const bf16x8*)(pga + 8);
      nb0 = *(const bf16x8*)(pgb);
      nb1 = *(const bf16x8*)(pgb + 8);
    }

    const bf16* as = &As[i & 1][0];
    const bf16* bs = &Bs[i & 1][0];
    // hoist ALL fragment loads before the WMMAs (hide ds latency)
    v16bf af0 = ld_frag(as + (wm * 32 + 0)  * 40, 40, lane);
    v16bf af1 = ld_frag(as + (wm * 32 + 16) * 40, 40, lane);
    v16bf bf0 = ld_frag(bs + (wn * 64 + 0)  * 40, 40, lane);
    v16bf bf1 = ld_frag(bs + (wn * 64 + 16) * 40, 40, lane);
    v16bf bf2 = ld_frag(bs + (wn * 64 + 32) * 40, 40, lane);
    v16bf bf3 = ld_frag(bs + (wn * 64 + 48) * 40, 40, lane);

    acc[0][0] = wmma_bf16(af0, bf0, acc[0][0]);
    acc[1][0] = wmma_bf16(af1, bf0, acc[1][0]);
    acc[0][1] = wmma_bf16(af0, bf1, acc[0][1]);
    acc[1][1] = wmma_bf16(af1, bf1, acc[1][1]);
    acc[0][2] = wmma_bf16(af0, bf2, acc[0][2]);
    acc[1][2] = wmma_bf16(af1, bf2, acc[1][2]);
    acc[0][3] = wmma_bf16(af0, bf3, acc[0][3]);
    acc[1][3] = wmma_bf16(af1, bf3, acc[1][3]);

    if (pf) {
      bf16* asn = &As[(i + 1) & 1][0];
      bf16* bsn = &Bs[(i + 1) & 1][0];
      *(bf16x8*)&asn[srow * 40 + co]     = na0;
      *(bf16x8*)&asn[srow * 40 + co + 8] = na1;
      *(bf16x8*)&bsn[srow * 40 + co]     = nb0;
      *(bf16x8*)&bsn[srow * 40 + co + 8] = nb1;
    }
  }

  const int halfo = (lane >> 4) << 3;
#pragma unroll
  for (int a = 0; a < 2; ++a)
#pragma unroll
    for (int bn = 0; bn < 4; ++bn)
#pragma unroll
      for (int r = 0; r < 8; ++r) {
        int mrow = m0 + wm * 32 + a * 16 + r + halfo;
        int ncol = n0 + wn * 64 + bn * 16 + (lane & 15);
        float v = acc[a][bn][r] + bias[ncol];
        if (do_gelu) v = 0.5f * v * (1.0f + erff(v * 0.70710678118f));
        if (resid) v += resid[(size_t)mrow * N + ncol];
        if (Cf) Cf[(size_t)mrow * N + ncol] = v;
        if (Cb) Cb[(size_t)mrow * N + ncol] = (bf16)v;
      }
}

// ---------------- flash attention (causal + pad mask) ----------------
// QKV: bf16 [M_TOK, 3072] fused buffer (Q | K | V per row).
// grid: (SEQ/64, BATCH*N_HEADS), 128 threads (4 waves x 16 q-rows)
__global__ __launch_bounds__(128) void attn_kernel(
    const bf16* __restrict__ QKV, const int* __restrict__ tokens,
    bf16* __restrict__ Ob) {
  __shared__ bf16 Kt[128 * 72];    // keys row-major [key][d]
  __shared__ bf16 Vt[64 * 136];    // V transposed  [d][key]
  __shared__ bf16 Pm[64 * 136];    // probabilities [qrow][key]
  __shared__ float padm[128];

  const int lane = threadIdx.x & 31;
  const int wave = threadIdx.x >> 5;         // 0..3
  const int b = blockIdx.y >> 4;
  const int h = blockIdx.y & 15;
  const int q0 = blockIdx.x * 64;
  const int halfo = (lane >> 4) << 3;

  const bf16* Qp = QKV + h * D_HEAD;                 // col 0..1023
  const bf16* Kp = QKV + D_MODEL + h * D_HEAD;       // col 1024..2047
  const bf16* Vp = QKV + 2 * D_MODEL + h * D_HEAD;   // col 2048..3071

  // Q fragments (held in VGPRs for the whole kernel)
  const bf16* qp = Qp + ((size_t)(b * SEQ + q0 + wave * 16)) * QKV_N;
  v16bf qf0 = ld_frag(qp + 0, QKV_N, lane);
  v16bf qf1 = ld_frag(qp + 32, QKV_N, lane);

  float m_s[8], l_s[8];
  v8f o[4];
#pragma unroll
  for (int r = 0; r < 8; ++r) { m_s[r] = -3.0e38f; l_s[r] = 0.f; }
#pragma unroll
  for (int dt = 0; dt < 4; ++dt)
#pragma unroll
    for (int r = 0; r < 8; ++r) o[dt][r] = 0.f;

  const int ktmax = (q0 + 63) >> 7;
  for (int kt = 0; kt <= ktmax; ++kt) {
    const int k0 = kt * 128;
    __syncthreads();
    {  // stage K, V(transposed), pad mask
      int r = threadIdx.x;  // one key row per thread
      const bf16* kp = Kp + ((size_t)(b * SEQ + k0 + r)) * QKV_N;
#pragma unroll
      for (int c = 0; c < 64; c += 8)
        *(bf16x8*)&Kt[r * 72 + c] = *(const bf16x8*)(kp + c);
      const bf16* vp = Vp + ((size_t)(b * SEQ + k0 + r)) * QKV_N;
#pragma unroll
      for (int c = 0; c < 64; ++c) Vt[c * 136 + r] = vp[c];
      padm[r] = (tokens[b * SEQ + k0 + r] == 0) ? NEGV : 0.0f;
    }
    __syncthreads();

    // S = (Q K^T) * 1/sqrt(d) + masks
    v8f sfr[8];
#pragma unroll
    for (int nt = 0; nt < 8; ++nt) {
      v8f s;
#pragma unroll
      for (int r = 0; r < 8; ++r) s[r] = 0.f;
      v16bf kf0 = ld_frag(&Kt[(nt * 16) * 72 + 0], 72, lane);
      v16bf kf1 = ld_frag(&Kt[(nt * 16) * 72 + 32], 72, lane);
      s = wmma_bf16(qf0, kf0, s);
      s = wmma_bf16(qf1, kf1, s);
      sfr[nt] = s;
    }

    float mnew[8];
#pragma unroll
    for (int r = 0; r < 8; ++r) mnew[r] = m_s[r];
#pragma unroll
    for (int nt = 0; nt < 8; ++nt) {
      int kg = k0 + nt * 16 + (lane & 15);
      float pm = padm[nt * 16 + (lane & 15)];
#pragma unroll
      for (int r = 0; r < 8; ++r) {
        int qg = q0 + wave * 16 + r + halfo;
        float v = sfr[nt][r] * 0.125f + pm + ((kg > qg) ? NEGV : 0.0f);
        sfr[nt][r] = v;
        mnew[r] = fmaxf(mnew[r], v);
      }
    }
#pragma unroll
    for (int r = 0; r < 8; ++r)
#pragma unroll
      for (int mk = 1; mk < 16; mk <<= 1)
        mnew[r] = fmaxf(mnew[r], __shfl_xor(mnew[r], mk, 16));

    float alpha[8], rsum[8];
#pragma unroll
    for (int r = 0; r < 8; ++r) {
      alpha[r] = __expf(m_s[r] - mnew[r]);
      m_s[r] = mnew[r];
      rsum[r] = 0.f;
    }
#pragma unroll
    for (int nt = 0; nt < 8; ++nt)
#pragma unroll
      for (int r = 0; r < 8; ++r) {
        float p = __expf(sfr[nt][r] - m_s[r]);
        rsum[r] += p;
        Pm[(wave * 16 + r + halfo) * 136 + nt * 16 + (lane & 15)] = (bf16)p;
      }
#pragma unroll
    for (int r = 0; r < 8; ++r) {
#pragma unroll
      for (int mk = 1; mk < 16; mk <<= 1)
        rsum[r] += __shfl_xor(rsum[r], mk, 16);
      l_s[r] = l_s[r] * alpha[r] + rsum[r];
    }
#pragma unroll
    for (int dt = 0; dt < 4; ++dt)
#pragma unroll
      for (int r = 0; r < 8; ++r) o[dt][r] *= alpha[r];

    // O += P V  (contraction over 128 keys = 4 WMMA k-steps)
#pragma unroll
    for (int dt = 0; dt < 4; ++dt)
#pragma unroll
      for (int kk = 0; kk < 4; ++kk) {
        v16bf pf = ld_frag(&Pm[(wave * 16) * 136 + kk * 32], 136, lane);
        v16bf vf = ld_frag(&Vt[(dt * 16) * 136 + kk * 32], 136, lane);
        o[dt] = wmma_bf16(pf, vf, o[dt]);
      }
  }

#pragma unroll
  for (int dt = 0; dt < 4; ++dt)
#pragma unroll
    for (int r = 0; r < 8; ++r) {
      float v = o[dt][r] / (l_s[r] + 1e-20f);
      int qg = q0 + wave * 16 + r + halfo;
      Ob[((size_t)(b * SEQ + qg)) * D_MODEL + h * D_HEAD + dt * 16 +
         (lane & 15)] = (bf16)v;
    }
}

// ---------------- host launcher ----------------
extern "C" void kernel_launch(void* const* d_in, const int* in_sizes, int n_in,
                              void* d_out, int out_size, void* d_ws,
                              size_t ws_size, hipStream_t stream) {
  (void)in_sizes; (void)n_in; (void)out_size; (void)ws_size;
  const int*   tokens = (const int*)d_in[0];
  const float* emb    = (const float*)d_in[1];
  const float* pos    = (const float*)d_in[2];
  const float* ln1g = (const float*)d_in[3];
  const float* ln1b = (const float*)d_in[4];
  const float* ln2g = (const float*)d_in[5];
  const float* ln2b = (const float*)d_in[6];
  const float* ln3g = (const float*)d_in[7];
  const float* ln3b = (const float*)d_in[8];
  const float* Wq = (const float*)d_in[9];
  const float* bq = (const float*)d_in[10];
  const float* Wk = (const float*)d_in[11];
  const float* bk = (const float*)d_in[12];
  const float* Wv = (const float*)d_in[13];
  const float* bv = (const float*)d_in[14];
  const float* Wo = (const float*)d_in[15];
  const float* bo = (const float*)d_in[16];
  const float* W1 = (const float*)d_in[17];
  const float* b1 = (const float*)d_in[18];
  const float* W2 = (const float*)d_in[19];
  const float* b2 = (const float*)d_in[20];

  char* ws = (char*)d_ws;
  size_t off = 0;
  auto alloc = [&](size_t bytes) -> void* {
    void* p = ws + off;
    off = (off + bytes + 255) & ~(size_t)255;
    return p;
  };
  const size_t MD = (size_t)M_TOK * D_MODEL;
  float* xf    = (float*)alloc(MD * 4);
  bf16*  xb    = (bf16*)alloc(MD * 2);
  bf16*  wqkvt = (bf16*)alloc((size_t)QKV_N * D_MODEL * 2);  // [3072,1024]
  bf16*  wot   = (bf16*)alloc((size_t)D_MODEL * D_MODEL * 2);
  bf16*  w1t   = (bf16*)alloc((size_t)D_MODEL * FFN * 2);
  bf16*  w2t   = (bf16*)alloc((size_t)FFN * D_MODEL * 2);
  float* bqkv  = (float*)alloc((size_t)QKV_N * 4);
  bf16*  qkv   = (bf16*)alloc((size_t)M_TOK * QKV_N * 2);    // [M,3072]
  bf16*  ab    = (bf16*)alloc(MD * 2);
  float* aof   = (float*)alloc(MD * 4);
  float* x2f   = (float*)alloc(MD * 4);
  bf16*  hin   = (bf16*)alloc(MD * 2);
  bf16*  hb    = (bf16*)alloc((size_t)M_TOK * FFN * 2);

  // 1) weight transpose + bf16 convert (Q|K|V stacked along N)
  dim3 tg(D_MODEL / 32, D_MODEL / 32);
  transw_kernel<<<tg, 256, 0, stream>>>(Wq, wqkvt, D_MODEL, D_MODEL);
  transw_kernel<<<tg, 256, 0, stream>>>(Wk, wqkvt + (size_t)D_MODEL * D_MODEL, D_MODEL, D_MODEL);
  transw_kernel<<<tg, 256, 0, stream>>>(Wv, wqkvt + (size_t)2 * D_MODEL * D_MODEL, D_MODEL, D_MODEL);
  transw_kernel<<<tg, 256, 0, stream>>>(Wo, wot, D_MODEL, D_MODEL);
  transw_kernel<<<dim3(FFN / 32, D_MODEL / 32), 256, 0, stream>>>(W1, w1t, D_MODEL, FFN);
  transw_kernel<<<dim3(D_MODEL / 32, FFN / 32), 256, 0, stream>>>(W2, w2t, FFN, D_MODEL);
  pack3_kernel<<<QKV_N / 256, 256, 0, stream>>>(bq, bk, bv, bqkv);

  // 2) embedding + positional + LN1
  embed_ln_kernel<<<M_TOK, 256, 0, stream>>>(tokens, emb, pos, ln1g, ln1b, xf, xb);

  // 3) fused QKV projection (bf16 output [M, 3072])
  gemm_bf16_kernel<<<dim3(QKV_N / 128, M_TOK / 128), 256, 0, stream>>>(
      xb, wqkvt, bqkv, nullptr, nullptr, qkv, M_TOK, QKV_N, D_MODEL, 0);

  // 4) attention
  attn_kernel<<<dim3(SEQ / 64, BATCH * N_HEADS), 128, 0, stream>>>(qkv, tokens, ab);

  // 5) output projection (fp32 for LN2)
  gemm_bf16_kernel<<<dim3(D_MODEL / 128, M_TOK / 128), 256, 0, stream>>>(
      ab, wot, bo, nullptr, aof, nullptr, M_TOK, D_MODEL, D_MODEL, 0);

  // 6) x2 = x + LN2(attn_out); hin = LN3(x2)
  ln2ln3_kernel<<<M_TOK, 256, 0, stream>>>(xf, aof, ln2g, ln2b, ln3g, ln3b, x2f, hin);

  // 7) FFN
  gemm_bf16_kernel<<<dim3(FFN / 128, M_TOK / 128), 256, 0, stream>>>(
      hin, w1t, b1, nullptr, nullptr, hb, M_TOK, FFN, D_MODEL, 1);
  gemm_bf16_kernel<<<dim3(D_MODEL / 128, M_TOK / 128), 256, 0, stream>>>(
      hb, w2t, b2, x2f, (float*)d_out, nullptr, M_TOK, D_MODEL, FFN, 0);
}
